// CrossAttention_43851616092336
// MI455X (gfx1250) — matrix operand order repaired and checked by reading, not compiled, not run on hardware
//
#include <hip/hip_runtime.h>

typedef _Float16 half_t;
typedef __attribute__((ext_vector_type(16))) _Float16 v16h;
typedef __attribute__((ext_vector_type(8)))  _Float16 v8h;
typedef __attribute__((ext_vector_type(8)))  float    v8f;
typedef __attribute__((ext_vector_type(4)))  float    v4f;
typedef __attribute__((ext_vector_type(4)))  int      v4i;

#define D_MODEL  1024
#define N_HEADS  16
#define HEAD_DIM 64
#define BATCH    8
#define P_LEN    512
#define S_LEN    2048
#define SBLK     32     // keys per attention inner block

// ---------------------------------------------------------------------------
// CDNA5 async global->LDS staging (ASYNCcnt path). The builtin takes
// v4i addrspace(1)* (global src) and v4i addrspace(3)* (LDS dst).
// ---------------------------------------------------------------------------
#if __has_builtin(__builtin_amdgcn_global_load_async_to_lds_b128)
#define HAVE_ASYNC_LDS 1
#else
#define HAVE_ASYNC_LDS 0
#endif

typedef __attribute__((address_space(1))) v4i* gv4i_p;
typedef __attribute__((address_space(3))) v4i* lv4i_p;

static __device__ inline void cp16_to_lds(half_t* l, const half_t* g) {
#if HAVE_ASYNC_LDS
  __builtin_amdgcn_global_load_async_to_lds_b128((gv4i_p)g, (lv4i_p)l, 0, 0);
#else
  *(v8h*)l = *(const v8h*)g;
#endif
}

static __device__ inline void wait_async_lds() {
#if HAVE_ASYNC_LDS
#if __has_builtin(__builtin_amdgcn_s_wait_asynccnt)
  __builtin_amdgcn_s_wait_asynccnt(0);
#else
  asm volatile("s_wait_asynccnt 0" ::: "memory");
#endif
#endif
}

// ---------------------------------------------------------------------------
// WMMA helpers (gfx1250: V_WMMA_F32_16X16X32_F16, wave32)
// A frag (16x32 f16): lane<16 row M=lane, k = 0..15; lane>=16 k = 16..31
// B frag (32x16 f16): lane<16 col N=lane, k = 0..15; lane>=16 k = 16..31
// C/D (16x16 f32): lane = column N (mod 16), vgpr r = row M (lanes>=16: M=8+r)
// ---------------------------------------------------------------------------
static __device__ inline v8f wmma16(v16h a, v16h b, v8f c) {
  return __builtin_amdgcn_wmma_f32_16x16x32_f16(
      false, a, false, b, (short)0, c, false, false);
}

static __device__ inline v8f zero8() {
  v8f z;
#pragma unroll
  for (int r = 0; r < 8; ++r) z[r] = 0.0f;
  return z;
}

static __device__ inline v16h load_a_h(const half_t* __restrict__ p) {
  v8h lo = *(const v8h*)(p);
  v8h hi = *(const v8h*)(p + 8);
  v16h a;
#pragma unroll
  for (int e = 0; e < 8; ++e) { a[e] = lo[e]; a[8 + e] = hi[e]; }
  return a;
}

static __device__ inline v16h load_a_f32(const float* __restrict__ p) {
  v4f x0 = *(const v4f*)(p);
  v4f x1 = *(const v4f*)(p + 4);
  v4f x2 = *(const v4f*)(p + 8);
  v4f x3 = *(const v4f*)(p + 12);
  v16h a;
#pragma unroll
  for (int e = 0; e < 4; ++e) {
    a[e]      = (half_t)x0[e];
    a[4 + e]  = (half_t)x1[e];
    a[8 + e]  = (half_t)x2[e];
    a[12 + e] = (half_t)x3[e];
  }
  return a;
}

// ---------------------------------------------------------------------------
// Projection GEMM:  Y = X(f32)[M,1024] @ Wh(f16)[1024,1024]^T + bias
// Block 256 thr (8 waves, 4m x 2n), block tile 128x128, wave tile 32x64
// (2 A-frags x 4 B-frags -> 8 WMMAs per 32-deep k-step).
// T, MODE compile-time: MODE 0 -> [b,h,t,64] (Q,K); MODE 1 -> [b,h,64,t] (V^T)
// ---------------------------------------------------------------------------
template <int T, int MODE>
__global__ __launch_bounds__(256) void proj_gemm_kernel(
    const float* __restrict__ X, const half_t* __restrict__ Wh,
    const float* __restrict__ bias, half_t* __restrict__ Y) {
  const int lane = threadIdx.x & 31;
  const int wave = threadIdx.x >> 5;
  const int lm   = lane & 15;
  const int hiL  = lane >> 4;
  const int kb   = hiL * 16;
  const int bkb  = hiL * 16;
  const int m0   = blockIdx.y * 128 + (wave >> 1) * 32;
  const int n0   = blockIdx.x * 128 + (wave & 1) * 64;

  v8f acc[2][4];
#pragma unroll
  for (int i = 0; i < 2; ++i)
#pragma unroll
    for (int j = 0; j < 4; ++j) acc[i][j] = zero8();

  const float*  xr0 = X + (size_t)(m0 + lm) * D_MODEL + kb;
  const float*  xr1 = X + (size_t)(m0 + 16 + lm) * D_MODEL + kb;
  const half_t* wr[4];
#pragma unroll
  for (int j = 0; j < 4; ++j)
    wr[j] = Wh + (size_t)(n0 + j * 16 + lm) * D_MODEL + bkb;

  for (int kk = 0; kk < D_MODEL; kk += 32) {
    if (kk + 64 < D_MODEL) {
      __builtin_prefetch(xr0 + kk + 64, 0, 1);
      __builtin_prefetch(xr1 + kk + 64, 0, 1);
#pragma unroll
      for (int j = 0; j < 4; ++j) __builtin_prefetch(wr[j] + kk + 64, 0, 1);
    }
    v16h a0 = load_a_f32(xr0 + kk);
    v16h a1 = load_a_f32(xr1 + kk);
#pragma unroll
    for (int j = 0; j < 4; ++j) {
      v16h bj = *(const v16h*)(wr[j] + kk);
      acc[0][j] = wmma16(a0, bj, acc[0][j]);
      acc[1][j] = wmma16(a1, bj, acc[1][j]);
    }
  }

#pragma unroll
  for (int j = 0; j < 4; ++j) {
    const int   n  = n0 + j * 16 + lm;
    const float bn = bias[n];
    const int   h  = n >> 6;
    const int   d  = n & 63;
#pragma unroll
    for (int i = 0; i < 2; ++i) {
#pragma unroll
      for (int r = 0; r < 8; ++r) {
        const int m = m0 + i * 16 + hiL * 8 + r;
        const int b = m / T;        // power-of-two constant -> shift
        const int t = m - b * T;
        const float v = acc[i][j][r] + bn;
        size_t idx;
        if (MODE == 0)
          idx = (((size_t)(b * N_HEADS + h)) * (size_t)T + t) * HEAD_DIM + d;
        else
          idx = (((size_t)(b * N_HEADS + h)) * HEAD_DIM + d) * (size_t)T + t;
        Y[idx] = (half_t)v;
      }
    }
  }
}

// ---------------------------------------------------------------------------
// Output GEMM:  Out(f32) = Ctx(f16)[4096,1024] @ Wo(f16)[1024,1024]^T + bias
// Same 128x128 block tile / 32x64 wave tile.
// ---------------------------------------------------------------------------
__global__ __launch_bounds__(256) void out_gemm_kernel(
    const half_t* __restrict__ Ctx, const half_t* __restrict__ Wh,
    const float* __restrict__ bias, float* __restrict__ Out) {
  const int lane = threadIdx.x & 31;
  const int wave = threadIdx.x >> 5;
  const int lm   = lane & 15;
  const int hiL  = lane >> 4;
  const int kb   = hiL * 16;
  const int bkb  = hiL * 16;
  const int m0   = blockIdx.y * 128 + (wave >> 1) * 32;
  const int n0   = blockIdx.x * 128 + (wave & 1) * 64;

  v8f acc[2][4];
#pragma unroll
  for (int i = 0; i < 2; ++i)
#pragma unroll
    for (int j = 0; j < 4; ++j) acc[i][j] = zero8();

  const half_t* ar0 = Ctx + (size_t)(m0 + lm) * D_MODEL + kb;
  const half_t* ar1 = Ctx + (size_t)(m0 + 16 + lm) * D_MODEL + kb;
  const half_t* wr[4];
#pragma unroll
  for (int j = 0; j < 4; ++j)
    wr[j] = Wh + (size_t)(n0 + j * 16 + lm) * D_MODEL + bkb;

  for (int kk = 0; kk < D_MODEL; kk += 32) {
    if (kk + 64 < D_MODEL) {
      __builtin_prefetch(ar0 + kk + 64, 0, 1);
      __builtin_prefetch(ar1 + kk + 64, 0, 1);
#pragma unroll
      for (int j = 0; j < 4; ++j) __builtin_prefetch(wr[j] + kk + 64, 0, 1);
    }
    v16h a0 = load_a_h(ar0 + kk);
    v16h a1 = load_a_h(ar1 + kk);
#pragma unroll
    for (int j = 0; j < 4; ++j) {
      v16h bj = *(const v16h*)(wr[j] + kk);
      acc[0][j] = wmma16(a0, bj, acc[0][j]);
      acc[1][j] = wmma16(a1, bj, acc[1][j]);
    }
  }

#pragma unroll
  for (int j = 0; j < 4; ++j) {
    const int   n  = n0 + j * 16 + lm;
    const float bn = bias[n];
#pragma unroll
    for (int i = 0; i < 2; ++i) {
#pragma unroll
      for (int r = 0; r < 8; ++r) {
        const int m = m0 + i * 16 + hiL * 8 + r;
        Out[(size_t)m * D_MODEL + n] = acc[i][j][r] + bn;
      }
    }
  }
}

// ---------------------------------------------------------------------------
// Flash attention: 8 waves/block, one 16-query tile per wave, all waves share
// double-buffered K/V tiles staged global->LDS via the CDNA5 async path.
//   S^T = K @ Q^T (WMMA), online softmax in a lane pair (1 shfl_xor),
//   P re-fragmented through per-wave LDS, O^T += V^T @ P^T (WMMA).
// ---------------------------------------------------------------------------
__global__ __launch_bounds__(256) void attn_kernel(
    const half_t* __restrict__ Qh, const half_t* __restrict__ Kh,
    const half_t* __restrict__ Vt, half_t* __restrict__ Ctx) {
  __shared__ half_t Kbuf[2][SBLK * HEAD_DIM];   // [2][32][64]  8 KB
  __shared__ half_t Vbuf[2][HEAD_DIM * SBLK];   // [2][64][32]  8 KB
  __shared__ half_t Pbuf[8][16 * SBLK];         // per-wave P    8 KB

  const int tid  = threadIdx.x;
  const int lane = tid & 31;
  const int wave = tid >> 5;
  const int lm   = lane & 15;
  const int hiL  = lane >> 4;
  const int kb   = hiL * 16;
  const int bkb  = hiL * 16;
  const int h = blockIdx.y, b = blockIdx.z;
  const int qt = blockIdx.x * 8 + wave;               // this wave's query tile
  const size_t bh = (size_t)b * N_HEADS + h;

  const half_t* Kg = Kh + bh * (size_t)S_LEN * HEAD_DIM;
  const half_t* Vg = Vt + bh * (size_t)HEAD_DIM * S_LEN;
  half_t* Pm = Pbuf[wave];

  // Q^T B-fragments (k = d); lane column = query row m.
  const half_t* qrow = Qh + (bh * P_LEN + (size_t)(qt * 16 + lm)) * HEAD_DIM;
  const v16h qb0 = *(const v16h*)(qrow + bkb);
  const v16h qb1 = *(const v16h*)(qrow + 32 + bkb);

  // stage first K/V block: K tile is 4KB contiguous; V tile is 64 rows x 64B
  {
    cp16_to_lds(&Kbuf[0][tid * 8], Kg + tid * 8);
    const int r = tid >> 2, c = (tid & 3) * 8;
    cp16_to_lds(&Vbuf[0][r * SBLK + c], Vg + (size_t)r * S_LEN + c);
  }

  float mrun = -1e30f, lrun = 0.0f;
  v8f o[4];
#pragma unroll
  for (int t = 0; t < 4; ++t) o[t] = zero8();

  for (int sb = 0; sb < S_LEN; sb += SBLK) {
    const int buf = (sb >> 5) & 1;
    wait_async_lds();
    __syncthreads();                       // staged tiles visible to all waves
    if (sb + SBLK < S_LEN) {               // issue next block into other buffer
      const int nsb = sb + SBLK;
      cp16_to_lds(&Kbuf[buf ^ 1][tid * 8], Kg + (size_t)nsb * HEAD_DIM + tid * 8);
      const int r = tid >> 2, c = (tid & 3) * 8;
      cp16_to_lds(&Vbuf[buf ^ 1][r * SBLK + c], Vg + (size_t)r * S_LEN + nsb + c);
    }

    // ---- S^T tiles: [s x m] = K[s, d] @ Q^T[d, m] ----
    v8f s0 = zero8(), s1 = zero8();
    {
      const half_t* k0 = &Kbuf[buf][lm * HEAD_DIM];
      const half_t* k1 = &Kbuf[buf][(16 + lm) * HEAD_DIM];
      s0 = wmma16(load_a_h(k0 + kb),      qb0, s0);
      s0 = wmma16(load_a_h(k0 + 32 + kb), qb1, s0);
      s1 = wmma16(load_a_h(k1 + kb),      qb0, s1);
      s1 = wmma16(load_a_h(k1 + 32 + kb), qb1, s1);
    }

    // ---- online softmax for query m = lm (lane pair lm, lm+16) ----
    float p[16];
    float mx = -1e30f;
#pragma unroll
    for (int r = 0; r < 8; ++r) {
      p[r]     = s0[r] * 0.125f;    // 1/sqrt(64)
      p[8 + r] = s1[r] * 0.125f;
    }
#pragma unroll
    for (int e = 0; e < 16; ++e) mx = fmaxf(mx, p[e]);
    mx = fmaxf(mx, __shfl_xor(mx, 16, 32));
    const float mnew  = fmaxf(mrun, mx);
    const float alpha = __expf(mrun - mnew);
    float rs = 0.0f;
#pragma unroll
    for (int e = 0; e < 16; ++e) { p[e] = __expf(p[e] - mnew); rs += p[e]; }
    rs += __shfl_xor(rs, 16, 32);
    lrun = lrun * alpha + rs;
    mrun = mnew;
#pragma unroll
    for (int t = 0; t < 4; ++t)
#pragma unroll
      for (int r = 0; r < 8; ++r) o[t][r] *= alpha;

    // ---- stage P into per-wave LDS: Pm[m][s] ----
#pragma unroll
    for (int r = 0; r < 8; ++r) Pm[lm * SBLK + hiL * 8 + r]      = (half_t)p[r];
#pragma unroll
    for (int r = 0; r < 8; ++r) Pm[lm * SBLK + 16 + hiL * 8 + r] = (half_t)p[8 + r];
    __syncthreads();

    // ---- O^T[d, m] += V^T[d, s] @ P^T[s, m] ----
    const v16h pb = *(const v16h*)(&Pm[lm * SBLK + bkb]);
#pragma unroll
    for (int t = 0; t < 4; ++t) {
      o[t] = wmma16(load_a_h(&Vbuf[buf][(t * 16 + lm) * SBLK + kb]), pb, o[t]);
    }
    __syncthreads();
  }

  // ---- normalize and store ctx[b, p, h*64 + d] (contiguous over d) ----
  const float inv = 1.0f / lrun;
  const size_t cbase =
      ((size_t)b * P_LEN + (size_t)(qt * 16 + lm)) * D_MODEL + h * HEAD_DIM;
#pragma unroll
  for (int t = 0; t < 4; ++t) {
    v8h pk;
#pragma unroll
    for (int r = 0; r < 8; ++r) pk[r] = (half_t)(o[t][r] * inv);
    *(v8h*)(Ctx + cbase + t * 16 + hiL * 8) = pk;
  }
}

// ---------------------------------------------------------------------------
__global__ __launch_bounds__(256) void cvt_kernel(const float* __restrict__ in,
                                                  half_t* __restrict__ out, int n) {
  int i = blockIdx.x * blockDim.x + threadIdx.x;
  if (i < n) out[i] = (half_t)in[i];
}

// ---------------------------------------------------------------------------
extern "C" void kernel_launch(void* const* d_in, const int* in_sizes, int n_in,
                              void* d_out, int out_size, void* d_ws, size_t ws_size,
                              hipStream_t stream) {
  const float* ts  = (const float*)d_in[0];
  const float* llm = (const float*)d_in[1];
  const float* q_w = (const float*)d_in[2];
  const float* q_b = (const float*)d_in[3];
  const float* k_w = (const float*)d_in[4];
  const float* k_b = (const float*)d_in[5];
  const float* v_w = (const float*)d_in[6];
  const float* v_b = (const float*)d_in[7];
  const float* o_w = (const float*)d_in[8];
  const float* o_b = (const float*)d_in[9];
  float* out = (float*)d_out;

  char* ws = (char*)d_ws;
  const size_t MB = 1ull << 20;
  half_t* Wq  = (half_t*)(ws + 0 * MB);
  half_t* Wk  = (half_t*)(ws + 2 * MB);
  half_t* Wv  = (half_t*)(ws + 4 * MB);
  half_t* Wo  = (half_t*)(ws + 6 * MB);
  half_t* Qh  = (half_t*)(ws + 8 * MB);    // [8,16,512,64]
  half_t* Kh  = (half_t*)(ws + 16 * MB);   // [8,16,2048,64]
  half_t* Vt  = (half_t*)(ws + 48 * MB);   // [8,16,64,2048]
  half_t* Ctx = (half_t*)(ws + 80 * MB);   // [8,512,1024]

  const int nw = D_MODEL * D_MODEL;
  cvt_kernel<<<nw / 256, 256, 0, stream>>>(q_w, Wq, nw);
  cvt_kernel<<<nw / 256, 256, 0, stream>>>(k_w, Wk, nw);
  cvt_kernel<<<nw / 256, 256, 0, stream>>>(v_w, Wv, nw);
  cvt_kernel<<<nw / 256, 256, 0, stream>>>(o_w, Wo, nw);

  proj_gemm_kernel<P_LEN, 0><<<dim3(D_MODEL / 128, (BATCH * P_LEN) / 128), 256, 0,
                               stream>>>(ts, Wq, q_b, Qh);
  proj_gemm_kernel<S_LEN, 0><<<dim3(D_MODEL / 128, (BATCH * S_LEN) / 128), 256, 0,
                               stream>>>(llm, Wk, k_b, Kh);
  proj_gemm_kernel<S_LEN, 1><<<dim3(D_MODEL / 128, (BATCH * S_LEN) / 128), 256, 0,
                               stream>>>(llm, Wv, v_b, Vt);

  attn_kernel<<<dim3(P_LEN / 16 / 8, N_HEADS, BATCH), 256, 0, stream>>>(
      Qh, Kh, Vt, Ctx);

  out_gemm_kernel<<<dim3(D_MODEL / 128, (BATCH * P_LEN) / 128), 256, 0, stream>>>(
      Ctx, Wo, o_b, out);
}
